// ContextualAttention_21861383536768
// MI455X (gfx1250) — matrix operand order repaired and checked by reading, not compiled, not run on hardware
//
#include <hip/hip_runtime.h>
#include <hip/hip_bf16.h>

#define EMB   1024
#define HEADS 16
#define HS    64
#define BB    4
#define ITOK  1024
#define CTOK  1024
#define TTOK  2048
#define INV_SCALE 0.17677669529663687f   /* 1/1024^0.25 */
#define NEGBIG   -103.27892990343185f    /* logf(1e-45) */

#ifndef __has_builtin
#define __has_builtin(x) 0
#endif
#if __has_builtin(__builtin_amdgcn_global_load_async_to_lds_b128) && \
    __has_builtin(__builtin_amdgcn_s_wait_asynccnt)
#define USE_ASYNC_LDS 1
#else
#define USE_ASYNC_LDS 0
#endif

typedef __attribute__((ext_vector_type(16))) __bf16 v16bf;
typedef __attribute__((ext_vector_type(8)))  __bf16 v8bf;
typedef __attribute__((ext_vector_type(8)))  float  v8f;
typedef __attribute__((ext_vector_type(4)))  int    v4i;

__device__ __forceinline__ unsigned short f2bf(float f) {
  unsigned u = __float_as_uint(f);
  u += 0x7FFFu + ((u >> 16) & 1u);
  return (unsigned short)(u >> 16);
}
__device__ __forceinline__ float bf2f(unsigned short h) {
  return __uint_as_float(((unsigned)h) << 16);
}
__device__ __forceinline__ v8f wmma_bf16(v16bf a, v16bf b, v8f c) {
  return __builtin_amdgcn_wmma_f32_16x16x32_bf16(false, a, false, b, (short)0, c,
                                                 false, false);
}
__device__ __forceinline__ v16bf pack16(v8bf lo, v8bf hi) {
  v16bf r;
#pragma unroll
  for (int i = 0; i < 8; ++i) { r[i] = lo[i]; r[8 + i] = hi[i]; }
  return r;
}
// A-fragment (16x32 bf16): lane<16 holds M=lane, K {0..15};
// lane>=16 holds M=lane-16, K {16..31} (kkA=8 -> offset 2*kkA).
__device__ __forceinline__ v16bf load_afrag(const unsigned short* row, int kkA) {
  v8bf lo = *(const v8bf*)(row + 2 * kkA);
  v8bf hi = *(const v8bf*)(row + 2 * kkA + 8);
  return pack16(lo, hi);
}

// ---------------- packing / conversion ----------------
__global__ void pack_xc(const float* __restrict__ x, const float* __restrict__ ctx,
                        unsigned short* __restrict__ xcb) {
  size_t i = (size_t)blockIdx.x * blockDim.x + threadIdx.x;
  if (i >= (size_t)BB * TTOK * EMB) return;
  size_t e = i % EMB;
  size_t t = (i / EMB) % TTOK;
  size_t b = i / ((size_t)EMB * TTOK);
  float v = (t < ITOK) ? x[(b * ITOK + t) * EMB + e]
                       : ctx[(b * CTOK + (t - ITOK)) * EMB + e];
  xcb[i] = f2bf(v);
}

__global__ void cvt_bf16(const float* __restrict__ in,
                         unsigned short* __restrict__ out, int n) {
  int i = blockIdx.x * blockDim.x + threadIdx.x;
  if (i < n) out[i] = f2bf(in[i]);
}

__global__ void mask_bias(const float* __restrict__ mq, const float* __restrict__ mc,
                          float* __restrict__ cb) {
  int i = blockIdx.x * blockDim.x + threadIdx.x;
  if (i >= BB * TTOK) return;
  int j = i % TTOK, b = i / TTOK;
  float m = (j < ITOK) ? mq[b * ITOK + j] : mc[b * CTOK + (j - ITOK)];
  cb[i] = __logf(m + 1e-45f);
}

// ------- WMMA GEMM: C = A(bf16 MxK) * W(bf16 NxK)^T ---------------------------
// WG = 4 waves, each wave computes 32x64; B tile (64n x 32k, 4KB) is staged in
// LDS once per k-step (double-buffered, async DMA one step ahead).
__global__ void __launch_bounds__(128)
gemm_bf16(const unsigned short* __restrict__ A, const unsigned short* __restrict__ W,
          int N, int K, size_t aBatchStride, size_t cBatchStride,
          float* __restrict__ outF, unsigned short* __restrict__ outB,
          const float* __restrict__ bias) {
  __shared__ __align__(128) unsigned short bbuf[2][64][32];
  const int lane = threadIdx.x, wv = threadIdx.y;
  const int tid  = wv * 32 + lane;
  const int m0 = blockIdx.y * 128 + wv * 32;
  const int n0 = blockIdx.x * 64;
  const unsigned short* Ab = A + (size_t)blockIdx.z * aBatchStride;
  const int kkA  = (lane < 16) ? 0 : 8;
  const int nc   = lane & 15;
  const int koB  = (lane < 16) ? 0 : 16;

  const unsigned short* arow0 = Ab + (size_t)(m0 + nc) * K;
  const unsigned short* arow1 = Ab + (size_t)(m0 + 16 + nc) * K;

  v8f acc0[4], acc1[4];
#pragma unroll
  for (int t = 0; t < 4; ++t)
#pragma unroll
    for (int r = 0; r < 8; ++r) { acc0[t][r] = 0.f; acc1[t][r] = 0.f; }

  // stage B tile for k-step k0 into bbuf[cur]; 128 threads x 2 x 16B = 4KB
  auto stage = [&](int cur, int k0) {
#pragma unroll
    for (int i = 0; i < 2; ++i) {
      int u = tid + i * 128;
      int row = u >> 2, c = (u & 3) * 8;
      const unsigned short* g = W + (size_t)(n0 + row) * K + k0 + c;
      unsigned short* l = &bbuf[cur][row][c];
#if USE_ASYNC_LDS
      __builtin_amdgcn_global_load_async_to_lds_b128((v4i*)g, (v4i*)l, 0, 0);
#else
      *(v8bf*)l = *(const v8bf*)g;
#endif
    }
  };

  const int nsteps = K / 32;
  stage(0, 0);
  for (int ks = 0; ks < nsteps; ++ks) {
    const int cur = ks & 1;
    const int k0 = ks * 32;
#if USE_ASYNC_LDS
    __builtin_amdgcn_s_wait_asynccnt(0);
#endif
    __syncthreads();                 // bbuf[cur] ready; prior compute done
    if (ks + 1 < nsteps) stage(cur ^ 1, (ks + 1) * 32);

    __builtin_prefetch(arow0 + k0 + 256, 0, 1);
    __builtin_prefetch(arow1 + k0 + 256, 0, 1);
    v16bf a0 = load_afrag(arow0 + k0, kkA);
    v16bf a1 = load_afrag(arow1 + k0, kkA);
#pragma unroll
    for (int t = 0; t < 4; ++t) {
      v16bf bt = *(const v16bf*)&bbuf[cur][t * 16 + nc][koB];
      acc0[t] = wmma_bf16(a0, bt, acc0[t]);
      acc1[t] = wmma_bf16(a1, bt, acc1[t]);
    }
  }

  const int moff = (lane < 16) ? 0 : 8;
#pragma unroll
  for (int t = 0; t < 4; ++t) {
    int n = n0 + t * 16 + nc;
    float bz = bias ? bias[n] : 0.f;
#pragma unroll
    for (int r = 0; r < 8; ++r) {
      size_t i0 = (size_t)blockIdx.z * cBatchStride +
                  (size_t)(m0 + r + moff) * N + n;
      size_t i1 = i0 + (size_t)16 * N;
      float v0 = acc0[t][r] + bz, v1 = acc1[t][r] + bz;
      if (outF) { outF[i0] = v0; outF[i1] = v1; }
      else      { outB[i0] = f2bf(v0); outB[i1] = f2bf(v1); }
    }
  }
}

// ------------- per-head LayerNorm + scale, row-major -> head-major bf16 -------
__global__ void __launch_bounds__(32)
ln_head(const unsigned short* __restrict__ src, unsigned short* __restrict__ dst,
        const float* __restrict__ lw, const float* __restrict__ lb, int rows) {
  const int h = blockIdx.x % HEADS, i = blockIdx.x / HEADS, b = blockIdx.y;
  const int lane = threadIdx.x;
  size_t srow = ((size_t)b * rows + i) * EMB + h * HS;
  float x0 = bf2f(src[srow + lane]);
  float x1 = bf2f(src[srow + lane + 32]);
  float s = x0 + x1, sq = x0 * x0 + x1 * x1;
#pragma unroll
  for (int m = 16; m >= 1; m >>= 1) {
    s  += __shfl_xor(s,  m, 32);
    sq += __shfl_xor(sq, m, 32);
  }
  float mu  = s * (1.f / 64.f);
  float var = sq * (1.f / 64.f) - mu * mu;
  float rstd = rsqrtf(var + 1e-5f);
  size_t drow = (((size_t)(b * HEADS + h)) * rows + i) * HS;
  dst[drow + lane]      = f2bf(((x0 - mu) * rstd * lw[lane]      + lb[lane])      * INV_SCALE);
  dst[drow + lane + 32] = f2bf(((x1 - mu) * rstd * lw[lane + 32] + lb[lane + 32]) * INV_SCALE);
}

// ------------- V reorder: (b,j,h*64+d) -> (b,h,d,j) ----------------
__global__ void v_trans(const unsigned short* __restrict__ vraw,
                        unsigned short* __restrict__ vT) {
  size_t i = (size_t)blockIdx.x * blockDim.x + threadIdx.x;
  if (i >= (size_t)BB * TTOK * EMB) return;
  int e = (int)(i % EMB);
  size_t j = (i / EMB) % TTOK, b = i / ((size_t)EMB * TTOK);
  int h = e / HS, d = e % HS;
  vT[(((size_t)(b * HEADS + h)) * HS + d) * TTOK + j] = vraw[i];
}

// ------------- flash attention: one wave = 16 query rows x full head ----------
__global__ void __launch_bounds__(128)
flash_attn(const unsigned short* __restrict__ qh, const unsigned short* __restrict__ kh,
           const unsigned short* __restrict__ vT, const float* __restrict__ maskq,
           const float* __restrict__ cbias, unsigned short* __restrict__ attnout) {
  __shared__ __align__(128) unsigned short pbuf[4][16][64];
  const int lane = threadIdx.x, wv = threadIdx.y;
  const int bh = blockIdx.y, b = bh / HEADS, h = bh % HEADS;
  const int i0 = blockIdx.x * 64 + wv * 16;
  const int nc   = lane & 15;
  const int kkA  = (lane < 16) ? 0 : 8;
  const int koB  = (lane < 16) ? 0 : 16;
  const int moff = (lane < 16) ? 0 : 8;

  const unsigned short* qrow = qh + ((size_t)bh * ITOK + i0 + nc) * HS;
  v16bf aq0 = load_afrag(qrow, kkA);
  v16bf aq1 = load_afrag(qrow + 32, kkA);

  bool rowon[8];
#pragma unroll
  for (int r = 0; r < 8; ++r)
    rowon[r] = (maskq[(size_t)b * ITOK + i0 + r + moff] != 0.f);

  float m_r[8], l_r[8];
  v8f acc[4];
#pragma unroll
  for (int r = 0; r < 8; ++r) { m_r[r] = -1e30f; l_r[r] = 0.f; }
#pragma unroll
  for (int t = 0; t < 4; ++t)
#pragma unroll
    for (int r = 0; r < 8; ++r) acc[t][r] = 0.f;

  const unsigned short* kbase = kh + (size_t)bh * TTOK * HS;
  const unsigned short* vbase = vT + (size_t)bh * HS * TTOK;
  const float* cb = cbias + (size_t)b * TTOK;

  for (int jc = 0; jc < TTOK; jc += 64) {
    v8f s[4];
    float cbf[4];
#pragma unroll
    for (int f = 0; f < 4; ++f) {
      const unsigned short* kr = kbase + (size_t)(jc + f * 16 + nc) * HS + koB;
      v16bf bk0 = *(const v16bf*)kr;
      v16bf bk1 = *(const v16bf*)(kr + 32);
      v8f z;
#pragma unroll
      for (int i = 0; i < 8; ++i) z[i] = 0.f;
      z = wmma_bf16(aq0, bk0, z);
      z = wmma_bf16(aq1, bk1, z);
      s[f] = z;
      cbf[f] = cb[jc + f * 16 + nc];
    }
    // online softmax: reg r <-> row (r+moff); 16-lane shfl groups are exact rows
#pragma unroll
    for (int r = 0; r < 8; ++r) {
      float mx = -1e30f;
#pragma unroll
      for (int f = 0; f < 4; ++f) {
        s[f][r] += rowon[r] ? cbf[f] : NEGBIG;
        mx = fmaxf(mx, s[f][r]);
      }
#pragma unroll
      for (int m = 8; m >= 1; m >>= 1) mx = fmaxf(mx, __shfl_xor(mx, m, 32));
      float nm = fmaxf(m_r[r], mx);
      float sc = __expf(m_r[r] - nm);
      float rs = 0.f;
#pragma unroll
      for (int f = 0; f < 4; ++f) {
        float p = __expf(s[f][r] - nm);
        s[f][r] = p;
        rs += p;
      }
#pragma unroll
      for (int m = 8; m >= 1; m >>= 1) rs += __shfl_xor(rs, m, 32);
      l_r[r] = l_r[r] * sc + rs;
      m_r[r] = nm;
#pragma unroll
      for (int t = 0; t < 4; ++t) acc[t][r] *= sc;
    }
    // C-layout P -> A-fragment via wave-private LDS (DS ops in-order per wave)
#pragma unroll
    for (int f = 0; f < 4; ++f)
#pragma unroll
      for (int r = 0; r < 8; ++r)
        pbuf[wv][r + moff][f * 16 + nc] = f2bf(s[f][r]);
    const unsigned short* pr = &pbuf[wv][nc][0];
    v16bf ap0 = load_afrag(pr, kkA);
    v16bf ap1 = load_afrag(pr + 32, kkA);
#pragma unroll
    for (int t = 0; t < 4; ++t) {
      const unsigned short* vr = vbase + (size_t)(t * 16 + nc) * TTOK + jc + koB;
      v16bf bv0 = *(const v16bf*)vr;
      v16bf bv1 = *(const v16bf*)(vr + 32);
      acc[t] = wmma_bf16(ap0, bv0, acc[t]);
      acc[t] = wmma_bf16(ap1, bv1, acc[t]);
    }
  }
#pragma unroll
  for (int r = 0; r < 8; ++r) l_r[r] = 1.f / l_r[r];
#pragma unroll
  for (int t = 0; t < 4; ++t)
#pragma unroll
    for (int r = 0; r < 8; ++r) {
      int i = i0 + r + moff;
      attnout[((size_t)b * ITOK + i) * EMB + h * HS + t * 16 + nc] =
          f2bf(acc[t][r] * l_r[r]);
    }
}

// ------------------------------- launcher -------------------------------
extern "C" void kernel_launch(void* const* d_in, const int* in_sizes, int n_in,
                              void* d_out, int out_size, void* d_ws, size_t ws_size,
                              hipStream_t stream) {
  const float* x     = (const float*)d_in[0];
  const float* ctx   = (const float*)d_in[1];
  const float* mask  = (const float*)d_in[2];
  const float* cmask = (const float*)d_in[3];
  const float* Wq    = (const float*)d_in[4];
  const float* Wk    = (const float*)d_in[5];
  const float* Wv    = (const float*)d_in[6];
  const float* Wu    = (const float*)d_in[7];
  const float* bu    = (const float*)d_in[8];
  const float* qlnw  = (const float*)d_in[9];
  const float* qlnb  = (const float*)d_in[10];
  const float* klnw  = (const float*)d_in[11];
  const float* klnb  = (const float*)d_in[12];

  char* ws = (char*)d_ws;
  unsigned short* xcb  = (unsigned short*)ws; ws += (size_t)BB * TTOK * EMB * 2;
  unsigned short* wqb  = (unsigned short*)ws; ws += (size_t)EMB * EMB * 2;
  unsigned short* wkb  = (unsigned short*)ws; ws += (size_t)EMB * EMB * 2;
  unsigned short* wvb  = (unsigned short*)ws; ws += (size_t)EMB * EMB * 2;
  unsigned short* wub  = (unsigned short*)ws; ws += (size_t)EMB * EMB * 2;
  unsigned short* qraw = (unsigned short*)ws; ws += (size_t)BB * ITOK * EMB * 2;
  unsigned short* kraw = (unsigned short*)ws; ws += (size_t)BB * TTOK * EMB * 2;
  unsigned short* vraw = (unsigned short*)ws; ws += (size_t)BB * TTOK * EMB * 2;
  unsigned short* qhb  = (unsigned short*)ws; ws += (size_t)BB * ITOK * EMB * 2;
  unsigned short* khb  = (unsigned short*)ws; ws += (size_t)BB * TTOK * EMB * 2;
  unsigned short* vTb  = (unsigned short*)ws; ws += (size_t)BB * TTOK * EMB * 2;
  unsigned short* atto = (unsigned short*)ws; ws += (size_t)BB * ITOK * EMB * 2;
  float* cb = (float*)ws;

  const size_t nxc = (size_t)BB * TTOK * EMB;
  pack_xc<<<(unsigned)((nxc + 255) / 256), 256, 0, stream>>>(x, ctx, xcb);
  cvt_bf16<<<(EMB * EMB + 255) / 256, 256, 0, stream>>>(Wq, wqb, EMB * EMB);
  cvt_bf16<<<(EMB * EMB + 255) / 256, 256, 0, stream>>>(Wk, wkb, EMB * EMB);
  cvt_bf16<<<(EMB * EMB + 255) / 256, 256, 0, stream>>>(Wv, wvb, EMB * EMB);
  cvt_bf16<<<(EMB * EMB + 255) / 256, 256, 0, stream>>>(Wu, wub, EMB * EMB);
  mask_bias<<<(BB * TTOK + 255) / 256, 256, 0, stream>>>(mask, cmask, cb);

  dim3 blk(32, 4);
  // Q = x @ Wq^T  (x rows are the first ITOK rows of each batch in xcb)
  gemm_bf16<<<dim3(EMB / 64, ITOK / 128, BB), blk, 0, stream>>>(
      xcb, wqb, EMB, EMB, (size_t)TTOK * EMB, (size_t)ITOK * EMB,
      nullptr, qraw, nullptr);
  // K, V = xc @ W^T  (all B*T rows contiguous)
  gemm_bf16<<<dim3(EMB / 64, (BB * TTOK) / 128, 1), blk, 0, stream>>>(
      xcb, wkb, EMB, EMB, 0, 0, nullptr, kraw, nullptr);
  gemm_bf16<<<dim3(EMB / 64, (BB * TTOK) / 128, 1), blk, 0, stream>>>(
      xcb, wvb, EMB, EMB, 0, 0, nullptr, vraw, nullptr);

  ln_head<<<dim3(ITOK * HEADS, BB), 32, 0, stream>>>(qraw, qhb, qlnw, qlnb, ITOK);
  ln_head<<<dim3(TTOK * HEADS, BB), 32, 0, stream>>>(kraw, khb, klnw, klnb, TTOK);
  v_trans<<<(unsigned)((nxc + 255) / 256), 256, 0, stream>>>(vraw, vTb);

  flash_attn<<<dim3(ITOK / 64, BB * HEADS), blk, 0, stream>>>(
      qhb, khb, vTb, mask, cb, atto);

  // out = attn_out @ Wu^T + bu  -> fp32 d_out
  gemm_bf16<<<dim3(EMB / 64, (BB * ITOK) / 128, 1), blk, 0, stream>>>(
      atto, wub, EMB, EMB, 0, 0, (float*)d_out, nullptr, bu);

  (void)in_sizes; (void)n_in; (void)out_size; (void)ws_size;
}
